// SDPA_6305011990831
// MI455X (gfx1250) — matrix operand order repaired and checked by reading, not compiled, not run on hardware
//
#include <hip/hip_runtime.h>
#include <hip/hip_bf16.h>

typedef __attribute__((ext_vector_type(16))) _Float16 v16h;
typedef __attribute__((ext_vector_type(8)))  _Float16 h8;
typedef __attribute__((ext_vector_type(8)))  float    v8f;
typedef __attribute__((ext_vector_type(4)))  _Float16 h4;
typedef __attribute__((ext_vector_type(4)))  unsigned u32x4;
typedef __attribute__((ext_vector_type(8)))  int      i32x8;
typedef __attribute__((ext_vector_type(4)))  int      i32x4;

#define DIMN      1024
#define BS        2
#define SEQ       2048
#define NH        16
#define HD        64
#define ROWS      (BS*SEQ)          // 4096 tokens

#if defined(__has_builtin)
#if __has_builtin(__builtin_amdgcn_tensor_load_to_lds) && \
    __has_builtin(__builtin_amdgcn_s_wait_tensorcnt)
#define HAVE_TDM 1
#endif
#endif

// ---------------------------------------------------------------------------
// CDNA5 async global->LDS copy (ASYNCcnt) + wait helpers
// ---------------------------------------------------------------------------
__device__ __forceinline__ void async_copy_b128(unsigned lds_byte_addr,
                                                const void* sbase,
                                                unsigned gbyte_off) {
  asm volatile("global_load_async_to_lds_b128 %0, %1, %2 offset:0"
               :: "v"(lds_byte_addr), "v"(gbyte_off), "s"(sbase)
               : "memory");
}
__device__ __forceinline__ void wait_async0() {
  asm volatile("s_wait_asynccnt 0x0" ::: "memory");
}
__device__ __forceinline__ unsigned lds_addr(const void* p) {
  return (unsigned)(uintptr_t)p;    // LDS generic address: offset in low 32 bits
}

// ---------------------------------------------------------------------------
// TDM: load a 2D f16 tile [rows x 32] with LDS row pad 8 halves (ld = 40)
// D# per CDNA5 ISA 8.3/8.4: count=1, type=2(image), data_size=1(2B),
// pad_enable, pad_interval=3 (16 DWORDs), pad_amount=3 (4 DWORDs).
// Issued by a single wave; TENSORcnt drained before the block barrier.
// This toolchain's builtin takes 6 args (g0,g1,g2,g3,extra,cpol).
// ---------------------------------------------------------------------------
__device__ __forceinline__ void tdm_load_tile_f16(unsigned lds_byte,
                                                  const _Float16* gsrc,
                                                  int rows, int row_stride_elems) {
#ifdef HAVE_TDM
  const unsigned long long ga = (unsigned long long)(uintptr_t)gsrc;
  u32x4 g0;
  g0[0] = 1u;                                            // count=1, user mode
  g0[1] = lds_byte;                                      // lds_addr (bytes)
  g0[2] = (unsigned)(ga & 0xffffffffu);                  // global_addr[31:0]
  g0[3] = (unsigned)((ga >> 32) & 0x01ffffffu) | (2u << 30);  // addr[56:32]|type=2
  i32x8 g1;
  g1[0] = (1 << 16) | (1 << 20) | (3 << 22) | (3 << 25); // 2B elems, pad 4dw/16dw
  g1[1] = (row_stride_elems & 0xffff) << 16;             // tensor_dim0[15:0]
  g1[2] = ((row_stride_elems >> 16) & 0xffff) | ((rows & 0xffff) << 16); // dim0 hi | dim1 lo
  g1[3] = (32 << 16);                                    // dim1 hi=0 | tile_dim0=32
  g1[4] = rows & 0xffff;                                 // tile_dim1 | tile_dim2=0
  g1[5] = row_stride_elems;                              // dim0_stride[31:0]
  g1[6] = 0;                                             // stride hi | dim1_stride lo
  g1[7] = 0;
  i32x4 gz4 = {0, 0, 0, 0};
  i32x8 gz8 = {0, 0, 0, 0, 0, 0, 0, 0};
  __builtin_amdgcn_tensor_load_to_lds(g0, g1, gz4, gz4, gz8, 0);
  __builtin_amdgcn_s_wait_tensorcnt(0);
#else
  (void)lds_byte; (void)gsrc; (void)rows; (void)row_stride_elems;
#endif
}

// ---------------------------------------------------------------------------
// DPP16 butterfly reductions within 16-lane rows (no LDS traffic)
// xor1 = quad_perm(1,0,3,2)=0xB1, xor2 = quad_perm(2,3,0,1)=0x4E,
// xor7 = ROW_HALF_MIRROR (0x141), xor15 = ROW_MIRROR (0x140)
// ---------------------------------------------------------------------------
template <int CTRL>
__device__ __forceinline__ float dppf(float x) {
  return __int_as_float(
      __builtin_amdgcn_mov_dpp(__float_as_int(x), CTRL, 0xf, 0xf, true));
}
__device__ __forceinline__ float row16_max(float x) {
  x = fmaxf(x, dppf<0xB1>(x));
  x = fmaxf(x, dppf<0x4E>(x));
  x = fmaxf(x, dppf<0x141>(x));
  x = fmaxf(x, dppf<0x140>(x));
  return x;
}
__device__ __forceinline__ float row16_sum(float x) {
  x += dppf<0xB1>(x);
  x += dppf<0x4E>(x);
  x += dppf<0x141>(x);
  x += dppf<0x140>(x);
  return x;
}

// ---------------------------------------------------------------------------
// WMMA fragment loaders (layouts per CDNA5 ISA 7.12.2, wave32)
// ---------------------------------------------------------------------------
__device__ __forceinline__ v16h load_frag_a(const _Float16* __restrict__ s, int ld) {
  const int lane = threadIdx.x & 31;
  const int m  = lane & 15;
  const int kh = (lane >> 4) << 3;
  h8 lo = *(const h8*)(s + m * ld + kh);
  h8 hi = *(const h8*)(s + m * ld + 16 + kh);
  v16h a;
#pragma unroll
  for (int i = 0; i < 8; ++i) { a[i] = lo[i]; a[8 + i] = hi[i]; }
  return a;
}

__device__ __forceinline__ v16h load_frag_bn(const _Float16* __restrict__ s, int ld) {
  const int lane = threadIdx.x & 31;
  const int n  = lane & 15;
  const int k0 = (lane >> 4) << 3;
  h8 lo = *(const h8*)(s + n * ld + k0);
  h8 hi = *(const h8*)(s + n * ld + 16 + k0);
  v16h b;
#pragma unroll
  for (int i = 0; i < 8; ++i) { b[i] = lo[i]; b[8 + i] = hi[i]; }
  return b;
}

__device__ __forceinline__ v8f wmma16(v16h a, v16h b, v8f c) {
  return __builtin_amdgcn_wmma_f32_16x16x32_f16(false, a, false, b, (short)0, c,
                                                false, false);
}

// ---------------------------------------------------------------------------
// fp32 -> fp16 convert (vectorized)
// ---------------------------------------------------------------------------
__global__ void cvt_f32_f16(const float* __restrict__ in, _Float16* __restrict__ out) {
  const int i = (blockIdx.x * 256 + threadIdx.x) * 4;
  float4 v = *(const float4*)(in + i);
  h4 h; h.x = (_Float16)v.x; h.y = (_Float16)v.y; h.z = (_Float16)v.z; h.w = (_Float16)v.w;
  *(h4*)(out + i) = h;
}

// ---------------------------------------------------------------------------
// 1024x1024 transpose + convert: out[d*N+o] = (f16) in[o*N+d]
// ---------------------------------------------------------------------------
__global__ void transpose_cvt(const float* __restrict__ in, _Float16* __restrict__ out) {
  __shared__ float tile[32][33];
  int x = blockIdx.x * 32 + threadIdx.x;
  int y = blockIdx.y * 32 + threadIdx.y;
#pragma unroll
  for (int i = 0; i < 32; i += 8)
    tile[threadIdx.y + i][threadIdx.x] = in[(y + i) * DIMN + x];
  __syncthreads();
  x = blockIdx.y * 32 + threadIdx.x;
  y = blockIdx.x * 32 + threadIdx.y;
#pragma unroll
  for (int i = 0; i < 32; i += 8)
    out[(y + i) * DIMN + x] = (_Float16)tile[threadIdx.x][threadIdx.y + i];
}

// ---------------------------------------------------------------------------
// Tiled WMMA GEMM: C[M,N] = A[M,K] * B[K,N]   (A,B f16 row-major in global)
// block tile 128x128, 256 threads = 8 waves (4 M x 2 N), wave tile 32x64.
// A tile staged by the Tensor Data Mover (TDM, pad->lda 40); falls back to
// per-lane async-to-LDS. B tile transposed to n-major through VGPRs.
// ---------------------------------------------------------------------------
template <bool HALF_OUT>
__global__ void gemm_wmma(const _Float16* __restrict__ A, const _Float16* __restrict__ B,
                          void* __restrict__ Cout, int M, int N, int K) {
  __shared__ __align__(16) _Float16 as[128 * 40];   // [m][k] lda 40
  __shared__ __align__(16) _Float16 bs[128 * 40];   // [n][k] ldb 40 (transposed)
  const int tid  = threadIdx.x;
  const int wave = tid >> 5;
  const int lane = tid & 31;
  const int wm = wave & 3;
  const int wn = wave >> 2;
  const int bm = blockIdx.y * 128;
  const int bn = blockIdx.x * 128;

  v8f acc[2][4];
#pragma unroll
  for (int i = 0; i < 2; ++i)
#pragma unroll
    for (int t = 0; t < 4; ++t) acc[i][t] = (v8f)0.0f;

  for (int k0 = 0; k0 < K; k0 += 32) {
    __syncthreads();
#ifdef HAVE_TDM
    if (wave == 0) {   // one TDM descriptor moves the whole 128x32 A tile
      tdm_load_tile_f16(lds_addr(as), A + (size_t)bm * K + k0, 128, K);
    }
#else
    { // A tile 128x32: per-lane async DMA, 2x16B per thread
      const int r = tid >> 1, c = (tid & 1) * 16;
      const unsigned l = lds_addr(as + r * 40 + c);
      const unsigned g = (unsigned)(((size_t)(bm + r) * K + k0 + c) * 2);
      async_copy_b128(l, A, g);
      async_copy_b128(l + 16, A, g + 16);
    }
#endif
    { // B tile 32x128 -> n-major: 16 halves/thread, scatter b16 stores
      const int r = tid >> 3, c = (tid & 7) * 16;
      const _Float16* src = B + (size_t)(k0 + r) * N + bn + c;
      h8 v0 = *(const h8*)(src);
      h8 v1 = *(const h8*)(src + 8);
#pragma unroll
      for (int i = 0; i < 8; ++i) bs[(c + i) * 40 + r]     = v0[i];
#pragma unroll
      for (int i = 0; i < 8; ++i) bs[(c + 8 + i) * 40 + r] = v1[i];
    }
#ifndef HAVE_TDM
    wait_async0();
#endif
    __syncthreads();

    v16h af[2], bf[4];
#pragma unroll
    for (int i = 0; i < 2; ++i) af[i] = load_frag_a(as + (wm * 32 + i * 16) * 40, 40);
#pragma unroll
    for (int t = 0; t < 4; ++t) bf[t] = load_frag_bn(bs + (wn * 64 + t * 16) * 40, 40);
#pragma unroll
    for (int i = 0; i < 2; ++i)
#pragma unroll
      for (int t = 0; t < 4; ++t) acc[i][t] = wmma16(af[i], bf[t], acc[i][t]);
  }

  // epilogue (C/D layout: lane&15 = N, lane>>4 -> M base 0/8, VGPR j = M)
  const int n  = lane & 15;
  const int mb = (lane >> 4) * 8;
#pragma unroll
  for (int i = 0; i < 2; ++i)
#pragma unroll
    for (int t = 0; t < 4; ++t)
#pragma unroll
      for (int j = 0; j < 8; ++j) {
        const size_t row = bm + wm * 32 + i * 16 + mb + j;
        const size_t col = bn + wn * 64 + t * 16 + n;
        if (HALF_OUT) ((_Float16*)Cout)[row * N + col] = (_Float16)acc[i][t][j];
        else          ((float*)Cout)[row * N + col]    = acc[i][t][j];
      }
}

// ---------------------------------------------------------------------------
// RoPE in place on Q and K (fp16 storage, fp32 math)
// ---------------------------------------------------------------------------
__global__ void rope_kernel(_Float16* __restrict__ Q, _Float16* __restrict__ Kk) {
  const int i = blockIdx.x * 256 + threadIdx.x;   // ROWS*NH*32 threads
  const int f = i & 31;
  const int h = (i >> 5) & 15;
  const int row = i >> 9;
  const int s = row & (SEQ - 1);
  const float inv = __powf(10000.0f, -(float)(2 * f) / 64.0f);
  float sn, cs;
  __sincosf((float)s * inv, &sn, &cs);
  const size_t j1 = (size_t)row * DIMN + h * 64 + f;
  const size_t j2 = j1 + 32;
  float q1 = (float)Q[j1], q2 = (float)Q[j2];
  Q[j1] = (_Float16)(q1 * cs - q2 * sn);
  Q[j2] = (_Float16)(q2 * cs + q1 * sn);
  float k1 = (float)Kk[j1], k2 = (float)Kk[j2];
  Kk[j1] = (_Float16)(k1 * cs - k2 * sn);
  Kk[j2] = (_Float16)(k2 * cs + k1 * sn);
}

// ---------------------------------------------------------------------------
// Flash attention: grid (S/64, B*H), 128 threads = 4 waves, 16 q-rows/wave.
// Q/K tiles async-staged row-major; V tile transposed to d-major for P*V.
// Online softmax with DPP16 row reductions.
// ---------------------------------------------------------------------------
__global__ void attn_kernel(const _Float16* __restrict__ Qh, const _Float16* __restrict__ Kh,
                            const _Float16* __restrict__ Vh, _Float16* __restrict__ Oh) {
  __shared__ __align__(16) _Float16 qs[64 * 72];   // [q][d]
  __shared__ __align__(16) _Float16 ks[32 * 72];   // [key][d]  (n-major for QK^T)
  __shared__ __align__(16) _Float16 vs[64 * 40];   // [d][key]  (n-major for P*V)
  __shared__ __align__(16) _Float16 ps[4 * 16 * 40];

  const int tid  = threadIdx.x;
  const int wave = tid >> 5;
  const int lane = tid & 31;
  const int bh = blockIdx.y;
  const int b = bh >> 4, h = bh & 15;
  const int q0 = blockIdx.x * 64;
  const int rowbase = b * SEQ;
  const int colbase = h * 64;

  { // stage Q 64x64 via async DMA (4x16B per thread)
    const int r = tid >> 1, c = (tid & 1) * 32;
    const unsigned l = lds_addr(qs + r * 72 + c);
    const unsigned g = (unsigned)(((size_t)(rowbase + q0 + r) * DIMN + colbase + c) * 2);
#pragma unroll
    for (int i = 0; i < 4; ++i) async_copy_b128(l + 16 * i, Qh, g + 16 * i);
  }
  wait_async0();
  __syncthreads();

  const v16h aQ0 = load_frag_a(qs + (wave * 16) * 72, 72);        // d 0..31
  const v16h aQ1 = load_frag_a(qs + (wave * 16) * 72 + 32, 72);   // d 32..63

  float rmax[8], rsum[8];
  v8f oacc[4];
#pragma unroll
  for (int j = 0; j < 8; ++j) { rmax[j] = -1e30f; rsum[j] = 0.0f; }
#pragma unroll
  for (int t = 0; t < 4; ++t) oacc[t] = (v8f)0.0f;

  const float scl = 0.125f;              // 1/sqrt(64)
  _Float16* myps = ps + wave * 16 * 40;
  const int n = lane & 15;
  const int g16 = lane >> 4;

  for (int kb = 0; kb < SEQ; kb += 32) {
    __syncthreads();
    { // K block 32x64 row-major: async DMA (2x16B per thread)
      const int r = tid >> 2, c = (tid & 3) * 16;
      const unsigned l = lds_addr(ks + r * 72 + c);
      const unsigned g = (unsigned)(((size_t)(rowbase + kb + r) * DIMN + colbase + c) * 2);
      async_copy_b128(l, Kh, g);
      async_copy_b128(l + 16, Kh, g + 16);
    }
    { // V block 32x64 -> d-major transpose (16 halves/thread, b16 scatter)
      const int r = tid >> 2, c = (tid & 3) * 16;
      const _Float16* src = Vh + (size_t)(rowbase + kb + r) * DIMN + colbase + c;
      h8 v0 = *(const h8*)(src);
      h8 v1 = *(const h8*)(src + 8);
#pragma unroll
      for (int i = 0; i < 8; ++i) vs[(c + i) * 40 + r]     = v0[i];
#pragma unroll
      for (int i = 0; i < 8; ++i) vs[(c + 8 + i) * 40 + r] = v1[i];
    }
    wait_async0();
    __syncthreads();

    // scores: 16q x 32k, contraction over d=64 (2 WMMAs per 16-key subtile)
    v8f s0 = (v8f)0.0f, s1 = (v8f)0.0f;
    s0 = wmma16(aQ0, load_frag_bn(ks, 72), s0);
    s0 = wmma16(aQ1, load_frag_bn(ks + 32, 72), s0);
    s1 = wmma16(aQ0, load_frag_bn(ks + 16 * 72, 72), s1);
    s1 = wmma16(aQ1, load_frag_bn(ks + 16 * 72 + 32, 72), s1);

    float alpha[8];
#pragma unroll
    for (int j = 0; j < 8; ++j) {
      const float a = s0[j] * scl, c = s1[j] * scl;
      const float m = row16_max(fmaxf(a, c));
      const float nm = fmaxf(rmax[j], m);
      alpha[j] = __expf(rmax[j] - nm);
      rmax[j] = nm;
      const float p0 = __expf(a - nm);
      const float p1 = __expf(c - nm);
      s0[j] = p0; s1[j] = p1;
      rsum[j] = rsum[j] * alpha[j] + row16_sum(p0 + p1);
    }
#pragma unroll
    for (int t = 0; t < 4; ++t)
#pragma unroll
      for (int j = 0; j < 8; ++j) oacc[t][j] *= alpha[j];

    // restage P (16x32) into per-wave LDS, reload as A fragment
#pragma unroll
    for (int j = 0; j < 8; ++j) {
      myps[(g16 * 8 + j) * 40 + n]      = (_Float16)s0[j];
      myps[(g16 * 8 + j) * 40 + 16 + n] = (_Float16)s1[j];
    }
    asm volatile("s_wait_dscnt 0x0" ::: "memory");   // wave-local LDS RAW fence
    const v16h aP = load_frag_a(myps, 40);
#pragma unroll
    for (int t = 0; t < 4; ++t)
      oacc[t] = wmma16(aP, load_frag_bn(vs + (t * 16) * 40, 40), oacc[t]);
  }

  // normalize + store (f16)
#pragma unroll
  for (int j = 0; j < 8; ++j) {
    const float inv = 1.0f / rsum[j];
    const size_t row = rowbase + q0 + wave * 16 + g16 * 8 + j;
#pragma unroll
    for (int t = 0; t < 4; ++t)
      Oh[row * DIMN + colbase + t * 16 + n] = (_Float16)(oacc[t][j] * inv);
  }
}

// ---------------------------------------------------------------------------
// launch
// ---------------------------------------------------------------------------
extern "C" void kernel_launch(void* const* d_in, const int* in_sizes, int n_in,
                              void* d_out, int out_size, void* d_ws, size_t ws_size,
                              hipStream_t stream) {
  const float* x  = (const float*)d_in[0];
  const float* Wq = (const float*)d_in[1];
  const float* Wk = (const float*)d_in[2];
  const float* Wv = (const float*)d_in[3];
  const float* Wo = (const float*)d_in[4];

  _Float16* ws  = (_Float16*)d_ws;
  _Float16* xh  = ws;                        // 4096*1024
  _Float16* wqt = xh  + (size_t)ROWS * DIMN; // 1024*1024 each
  _Float16* wkt = wqt + (size_t)DIMN * DIMN;
  _Float16* wvt = wkt + (size_t)DIMN * DIMN;
  _Float16* wot = wvt + (size_t)DIMN * DIMN;
  _Float16* Qh  = wot + (size_t)DIMN * DIMN;
  _Float16* Kh  = Qh  + (size_t)ROWS * DIMN;
  _Float16* Vh  = Kh  + (size_t)ROWS * DIMN;
  _Float16* Oh  = Vh  + (size_t)ROWS * DIMN;

  // 1) convert inputs to f16 (weights transposed so GEMM B is row-major [K,N])
  cvt_f32_f16<<<(ROWS * DIMN) / 1024, 256, 0, stream>>>(x, xh);
  dim3 tb(32, 8), tg(DIMN / 32, DIMN / 32);
  transpose_cvt<<<tg, tb, 0, stream>>>(Wq, wqt);
  transpose_cvt<<<tg, tb, 0, stream>>>(Wk, wkt);
  transpose_cvt<<<tg, tb, 0, stream>>>(Wv, wvt);
  transpose_cvt<<<tg, tb, 0, stream>>>(Wo, wot);

  // 2) Q/K/V projections
  dim3 gg(DIMN / 128, ROWS / 128);
  gemm_wmma<true><<<gg, 256, 0, stream>>>(xh, wqt, Qh, ROWS, DIMN, DIMN);
  gemm_wmma<true><<<gg, 256, 0, stream>>>(xh, wkt, Kh, ROWS, DIMN, DIMN);
  gemm_wmma<true><<<gg, 256, 0, stream>>>(xh, wvt, Vh, ROWS, DIMN, DIMN);

  // 3) RoPE on Q and K
  rope_kernel<<<(ROWS * NH * 32) / 256, 256, 0, stream>>>(Qh, Kh);

  // 4) attention
  attn_kernel<<<dim3(SEQ / 64, BS * NH), 128, 0, stream>>>(Qh, Kh, Vh, Oh);

  // 5) output projection (fp32 out)
  gemm_wmma<false><<<gg, 256, 0, stream>>>(Oh, wot, (float*)d_out, ROWS, DIMN, DIMN);
}
